// HierAttNet_65197603554039
// MI455X (gfx1250) — matrix operand order repaired and checked by reading, not compiled, not run on hardware
//
#include <hip/hip_runtime.h>
#include <hip/hip_bf16.h>

typedef __attribute__((ext_vector_type(16))) _Float16 v16h;
typedef __attribute__((ext_vector_type(8)))  _Float16 v8h;
typedef __attribute__((ext_vector_type(8)))  float    v8f;
typedef __attribute__((ext_vector_type(4)))  unsigned int u32x4;
typedef __attribute__((ext_vector_type(8)))  unsigned int u32x8;

#define WG 256

// ---- WMMA helpers (gfx1250, wave32) -------------------------------------
__device__ __forceinline__ v16h load_a_frag(const _Float16* base, int stride,
                                            int lane, int kbase) {
  int row = lane & 15;
  int k0  = kbase + ((lane < 16) ? 0 : 16);
  const _Float16* p = base + row * stride + k0;
  v8h lo = *(const v8h*)(p);
  v8h hi = *(const v8h*)(p + 8);
  return __builtin_shufflevector(lo, hi, 0, 1, 2, 3, 4, 5, 6, 7,
                                 8, 9, 10, 11, 12, 13, 14, 15);
}

__device__ __forceinline__ v8f wmma_f16(v16h a, v16h b, v8f c) {
  return __builtin_amdgcn_wmma_f32_16x16x32_f16(false, a, false, b,
                                                (short)0, c, false, false);
}

// ---- TDM: 2D tile load (rows x tile_dw dwords) global -> LDS -------------
// Builds D# group0 (4 SGPRs) + group1 (8 SGPRs) per CDNA5 ISA and issues
// tensor_load_to_lds. data_size = 4 bytes; strides/dims in dword units.
__device__ __forceinline__ void tdm_load_2d_to_lds(unsigned lds_off,
                                                   const void* gptr,
                                                   unsigned tile_dw,
                                                   unsigned rows,
                                                   unsigned stride_dw) {
  unsigned long long ga = (unsigned long long)gptr;
  u32x4 g0;
  g0[0] = 1u;                                   // count=1, user descriptor
  g0[1] = lds_off;                              // lds_addr (bytes)
  g0[2] = (unsigned)(ga & 0xffffffffu);         // global_addr[31:0]
  g0[3] = (unsigned)((ga >> 32) & 0x01ffffffu)  // global_addr[56:32]
        | (2u << 30);                           // type = 2 ("image")
  // group1 bitfields (see CDNA5 ISA)
  unsigned long long q0 = (2ull << 16)                                  // data_size=4B
                        | ((unsigned long long)(stride_dw & 0xffffu) << 48); // tensor_dim0[15:0]
  unsigned long long q1 = ((unsigned long long)(stride_dw >> 16) & 0xffffu)  // tensor_dim0[31:16]
                        | ((unsigned long long)rows << 16)              // tensor_dim1
                        | ((unsigned long long)(tile_dw & 0xffffu) << 48); // tile_dim0
  unsigned long long q2 = ((unsigned long long)(rows & 0xffffu))        // tile_dim1
                        | ((unsigned long long)stride_dw << 32);        // dim0_stride[31:0]
  u32x8 g1;
  g1[0] = (unsigned)q0; g1[1] = (unsigned)(q0 >> 32);
  g1[2] = (unsigned)q1; g1[3] = (unsigned)(q1 >> 32);
  g1[4] = (unsigned)q2; g1[5] = (unsigned)(q2 >> 32);
  g1[6] = 0u;           g1[7] = 0u;             // dim0_stride[47:32]=0, dim1_stride=0
  asm volatile("tensor_load_to_lds %0, %1" :: "s"(g0), "s"(g1) : "memory");
}

// ---- Kernel 0: f32 -> padded f16 conversion ------------------------------
__global__ __launch_bounds__(WG) void cvt_pad_kernel(
    _Float16* __restrict__ dst, const float* __restrict__ src,
    long rows_src, int K, int KPAD, long total) {
  long idx = (long)blockIdx.x * WG + threadIdx.x;
  long stride = (long)gridDim.x * WG;
  for (; idx < total; idx += stride) {
    long r = idx / KPAD;
    int  k = (int)(idx - r * KPAD);
    float v = 0.0f;
    if (r < rows_src && k < K) v = src[r * K + k];
    dst[idx] = (_Float16)v;
  }
}

// ---- Kernel 1/5: input projection  GI = X @ W_ih^T + b_ih  (WMMA) -------
// src16: pre-converted f16 rows with stride KPAD (zero-padded).
// w16:   pre-converted f16 weights [320][KPAD] (rows 300..319 zero).
// N = 320 => 20 N-tiles, each wave-parity group owns exactly 10 (registers).
// B-slab per k-step comes in via one TDM tensor_load_to_lds from wave 0.
template <int K, int KPAD>
__global__ __launch_bounds__(WG) void proj_kernel(
    float* __restrict__ out,            // [rows][304] f32
    const _Float16* __restrict__ src16, // [V|M][KPAD] f16
    const int* __restrict__ ids,        // nullable; id = ids[b*ids_stride+wg]
    int ids_stride,
    const _Float16* __restrict__ w16,   // [320][KPAD] f16
    const float* __restrict__ b_ih)     // [300]
{
  __shared__ _Float16 lA[64 * KPAD];  // A tile (f16, pre-padded source)
  __shared__ _Float16 lB[320 * 32];   // W slab for one k-step, [n][k]
  const int wg   = blockIdx.x;
  const int tid  = threadIdx.x;
  const int lane = tid & 31;
  const int w    = tid >> 5;
  const int wm   = w >> 1;            // M-tile (0..3)
  const int wn   = w & 1;             // N parity: tiles wn, wn+2, ..., wn+18

  // A tile: pure f16 row copy (8B chunks), gather via ids if present.
  constexpr int RQ = KPAD / 4;        // 8-byte chunks per row
  for (int idx = tid; idx < 64 * RQ; idx += WG) {
    int r = idx / RQ;
    int c = idx - r * RQ;
    long rowb;
    if (ids) {
      int id = ids[r * ids_stride + wg];
      if (id < 0) id = 0;
      if (id > 100000) id = 100000;
      rowb = id;
    } else {
      rowb = (long)wg * 64 + r;
    }
    const unsigned long long* sp =
        (const unsigned long long*)(src16 + rowb * KPAD);
    ((unsigned long long*)lA)[(long)r * RQ + c] = sp[c];
  }

  const v8f vz = {0.f, 0.f, 0.f, 0.f, 0.f, 0.f, 0.f, 0.f};
  v8f acc[10];
#pragma unroll
  for (int i = 0; i < 10; ++i) acc[i] = vz;

  const unsigned lB_off = (unsigned)(unsigned long long)(uintptr_t)(void*)lB;

  constexpr int KSTEPS = KPAD >> 5;
  for (int kk = 0; kk < KSTEPS; ++kk) {
    __syncthreads();   // prior iteration's WMMA reads of lB complete
    if (w == 0) {      // one wave drives the DMA; others wait at barrier
      tdm_load_2d_to_lds(lB_off, (const void*)(w16 + kk * 32),
                         /*tile_dw=*/16, /*rows=*/320,
                         /*stride_dw=*/KPAD / 2);
      __builtin_amdgcn_s_wait_tensorcnt(0);
    }
    __syncthreads();   // lB visible to all waves

    v16h a = load_a_frag(lA + (wm * 16) * KPAD, KPAD, lane, kk * 32);
#pragma unroll
    for (int it = 0; it < 10; ++it) {
      int nt = wn + 2 * it;
      int n  = nt * 16 + (lane & 15);
      int koff = (lane < 16) ? 0 : 16;
      v16h b = *(const v16h*)(lB + n * 32 + koff);
      acc[it] = wmma_f16(a, b, acc[it]);
    }
  }

#pragma unroll
  for (int it = 0; it < 10; ++it) {
    int nt = wn + 2 * it;
    int n  = nt * 16 + (lane & 15);
    if (n < 300) {
      float bias = b_ih[n];
      int mbase = wm * 16 + ((lane < 16) ? 0 : 8);
#pragma unroll
      for (int r = 0; r < 8; ++r) {
        long gr = (long)wg * 64 + mbase + r;
        out[gr * 304 + n] = acc[it][r] + bias;
      }
    }
  }
}

// ---- Kernel 2/6: serial BiGRU recurrence (WMMA per step) -----------------
// grid = 2 (blockIdx.x = direction). State lives in LDS; GI streamed from L2
// with software prefetch of the next step's slab.
__global__ __launch_bounds__(WG) void bigru_kernel(
    float* __restrict__ fout,        // [nseg*Tw][64][100]
    const float* __restrict__ gi,    // [nseg*Tw*64][304]
    const float* __restrict__ w_hh,  // [2][150][50]
    const float* __restrict__ b_hh,  // [2][150]
    int nseg, int Tw) {
  __shared__ _Float16 sW[160 * 64];    // W_hh in [n][k], padded
  __shared__ _Float16 sH[64 * 64];     // h as f16 A-operand [b][k]
  __shared__ _Float16 sGH[64 * 160];   // gh result [b][n]
  __shared__ float    sHf[64 * 50];    // h master copy f32
  __shared__ float    sB[152];         // b_hh for this direction

  const int dir  = blockIdx.x;
  const int tid  = threadIdx.x;
  const int lane = tid & 31;
  const int w    = tid >> 5;
  const int wm   = w >> 1;   // M-tile 0..3
  const int wn   = w & 1;    // N parity; each wave owns 5 of 10 N-tiles

  for (int idx = tid; idx < 160 * 64; idx += WG) {
    int n = idx >> 6, k = idx & 63;
    float v = 0.0f;
    if (n < 150 && k < 50) v = w_hh[(dir * 150 + n) * 50 + k];
    sW[idx] = (_Float16)v;
  }
  for (int idx = tid; idx < 64 * 64; idx += WG) sH[idx] = (_Float16)0.0f;
  for (int idx = tid; idx < 64 * 50; idx += WG) sHf[idx] = 0.0f;
  for (int idx = tid; idx < 150; idx += WG) sB[idx] = b_hh[dir * 150 + idx];
  __syncthreads();

  const v8f vz = {0.f, 0.f, 0.f, 0.f, 0.f, 0.f, 0.f, 0.f};
  for (int seg = 0; seg < nseg; ++seg) {
    for (int i = 0; i < Tw; ++i) {
      int tc = (dir == 0) ? i : (Tw - 1 - i);
      long st = (long)seg * Tw + tc;

      // Prefetch next step's gate-input slab into cache (global_prefetch_b8)
      {
        int i2 = i + 1;
        int seg2 = seg;
        if (i2 >= Tw) { i2 = 0; seg2 = seg + 1; }
        if (seg2 < nseg) {
          int tc2 = (dir == 0) ? i2 : (Tw - 1 - i2);
          long st2 = (long)seg2 * Tw + tc2;
          const float* pg = gi + (st2 * 64) * 304 + dir * 150;
          for (int p = tid; p < 64 * 5; p += WG) {
            int b = p / 5, c = p - b * 5;
            __builtin_prefetch(pg + (long)b * 304 + c * 32, 0, 1);
          }
        }
      }

      // gh = h @ W_hh^T : M=64, N=160(150), K=64(50) -> 2 k-steps of 32
      v8f acc[5];
#pragma unroll
      for (int it = 0; it < 5; ++it) acc[it] = vz;
#pragma unroll
      for (int kk = 0; kk < 2; ++kk) {
        v16h a = load_a_frag(sH + (wm * 16) * 64, 64, lane, kk * 32);
#pragma unroll
        for (int it = 0; it < 5; ++it) {
          int nt = wn + 2 * it;
          int n  = nt * 16 + (lane & 15);
          int koff = kk * 32 + ((lane < 16) ? 0 : 16);
          v16h b = *(const v16h*)(sW + n * 64 + koff);
          acc[it] = wmma_f16(a, b, acc[it]);
        }
      }
#pragma unroll
      for (int it = 0; it < 5; ++it) {
        int nt = wn + 2 * it;
        int n  = nt * 16 + (lane & 15);
        int mbase = wm * 16 + ((lane < 16) ? 0 : 8);
#pragma unroll
        for (int r = 0; r < 8; ++r)
          sGH[(mbase + r) * 160 + n] = (_Float16)acc[it][r];
      }
      __syncthreads();

      // gates: r,z,n per (batch, unit); torch GRUCell math
      for (int p = tid; p < 64 * 50; p += WG) {
        int b = p / 50, j = p - b * 50;
        long gbase = (st * 64 + b) * 304 + dir * 150;
        float gir = gi[gbase + j];
        float giz = gi[gbase + 50 + j];
        float gin = gi[gbase + 100 + j];
        float ghr = (float)sGH[b * 160 + j]        + sB[j];
        float ghz = (float)sGH[b * 160 + 50 + j]   + sB[50 + j];
        float ghn = (float)sGH[b * 160 + 100 + j]  + sB[100 + j];
        float rg = 1.0f / (1.0f + __expf(-(gir + ghr)));
        float zg = 1.0f / (1.0f + __expf(-(giz + ghz)));
        float ng = tanhf(gin + rg * ghn);
        float hp = sHf[p];
        float hn = (1.0f - zg) * ng + zg * hp;
        sHf[p] = hn;
        sH[b * 64 + j] = (_Float16)hn;
        fout[(st * 64 + b) * 100 + dir * 50 + j] = hn;
      }
      __syncthreads();
    }
  }
}

// ---- Kernel 3/7: attention scores  s = tanh(tanh(f@W + b) @ ctx) ---------
__global__ __launch_bounds__(WG) void scores_kernel(
    float* __restrict__ scores, const float* __restrict__ fin,
    const float* __restrict__ W, const float* __restrict__ bias,
    const float* __restrict__ ctx, int R) {
  __shared__ float sW[100 * 100];
  __shared__ float sBi[100];
  __shared__ float sCx[100];
  int tid = threadIdx.x;
  for (int idx = tid; idx < 100 * 100; idx += WG) sW[idx] = W[idx];
  for (int idx = tid; idx < 100; idx += WG) { sBi[idx] = bias[idx]; sCx[idx] = ctx[idx]; }
  __syncthreads();
  int row = blockIdx.x * WG + tid;
  if (row < R) {
    float f[100];
    const float* fp = fin + (long)row * 100;
#pragma unroll
    for (int k = 0; k < 100; ++k) f[k] = fp[k];
    float s = 0.0f;
    for (int n = 0; n < 100; ++n) {
      float u = sBi[n];
#pragma unroll 4
      for (int k = 0; k < 100; ++k) u += f[k] * sW[k * 100 + n];
      s += tanhf(u) * sCx[n];
    }
    scores[row] = tanhf(s);
  }
}

// ---- Kernel 4/8: softmax over BATCH axis (per t), weighted sum over T ----
__global__ __launch_bounds__(WG) void attnsum_kernel(
    float* __restrict__ out, const float* __restrict__ fin,
    const float* __restrict__ scores, int T) {
  __shared__ float sA[50 * 64];
  int seg = blockIdx.x, tid = threadIdx.x;
  for (int idx = tid; idx < T * 64; idx += WG)
    sA[idx] = scores[(long)seg * T * 64 + idx];
  __syncthreads();
  if (tid < T) {
    float m = -1e30f;
    for (int b = 0; b < 64; ++b) m = fmaxf(m, sA[tid * 64 + b]);
    float sum = 0.0f;
    for (int b = 0; b < 64; ++b) {
      float e = __expf(sA[tid * 64 + b] - m);
      sA[tid * 64 + b] = e;
      sum += e;
    }
    float inv = 1.0f / sum;
    for (int b = 0; b < 64; ++b) sA[tid * 64 + b] *= inv;
  }
  __syncthreads();
  for (int p = tid; p < 64 * 100; p += WG) {
    int b = p / 100, n = p - b * 100;
    float o = 0.0f;
    for (int t = 0; t < T; ++t)
      o += fin[(((long)seg * T + t) * 64 + b) * 100 + n] * sA[t * 64 + b];
    out[((long)seg * 64 + b) * 100 + n] = o;
  }
}

// ---- Kernel 9: final FC --------------------------------------------------
__global__ __launch_bounds__(WG) void fc_kernel(
    float* __restrict__ out, const float* __restrict__ doc,
    const float* __restrict__ fw, const float* __restrict__ fb) {
  int tid = threadIdx.x;
  for (int p = tid; p < 64 * 14; p += WG) {
    int b = p / 14, c = p - b * 14;
    float o = fb[c];
#pragma unroll 4
    for (int k = 0; k < 100; ++k) o += doc[b * 100 + k] * fw[c * 100 + k];
    out[p] = o;
  }
}

extern "C" void kernel_launch(void* const* d_in, const int* in_sizes, int n_in,
                              void* d_out, int out_size, void* d_ws, size_t ws_size,
                              hipStream_t stream) {
  (void)in_sizes; (void)n_in; (void)out_size; (void)ws_size;
  const int*   ids   = (const int*)  d_in[0];   // [64][30][50]
  const float* embed = (const float*)d_in[1];   // [100001][200]
  const float* wihw  = (const float*)d_in[2];   // [2][150][200]
  const float* whhw  = (const float*)d_in[3];   // [2][150][50]
  const float* bihw  = (const float*)d_in[4];   // [2][150]
  const float* bhhw  = (const float*)d_in[5];   // [2][150]
  const float* wWT   = (const float*)d_in[6];   // [100][100]
  const float* wB    = (const float*)d_in[7];   // [100]
  const float* wC    = (const float*)d_in[8];   // [100]
  const float* wihs  = (const float*)d_in[9];   // [2][150][100]
  const float* whhs  = (const float*)d_in[10];  // [2][150][50]
  const float* bihs  = (const float*)d_in[11];  // [2][150]
  const float* bhhs  = (const float*)d_in[12];  // [2][150]
  const float* sWT   = (const float*)d_in[13];  // [100][100]
  const float* sBi   = (const float*)d_in[14];  // [100]
  const float* sC    = (const float*)d_in[15];  // [100]
  const float* fcw   = (const float*)d_in[16];  // [14][100]
  const float* fcb   = (const float*)d_in[17];  // [14]
  float* out = (float*)d_out;                   // [64][14]

  // Workspace layout: f32 region then f16 region (~205 MB total)
  float* ws    = (float*)d_ws;
  float* GIw   = ws;                           // 96000 * 304
  float* foutw = GIw   + 96000L * 304;         // 1500*64*100
  float* scw   = foutw + 1500L * 64 * 100;     // 96000
  float* sseq  = scw   + 96000L;               // 30*64*100
  float* GIs   = sseq  + 30L * 64 * 100;       // 1920 * 304
  float* fouts = GIs   + 1920L * 304;          // 30*64*100
  float* scs   = fouts + 30L * 64 * 100;       // 1920
  float* doc   = scs   + 1920L;                // 64*100
  _Float16* h16 = (_Float16*)(doc + 6400L);    // f16 region (64B aligned)
  _Float16* E16  = h16;                        // 100001*224
  _Float16* W16w = E16  + 100001L * 224;       // 320*224
  _Float16* W16s = W16w + 320L * 224;          // 320*128
  _Float16* S16  = W16s + 320L * 128;          // 1920*128

  // Pre-convert operands to padded f16 (once, bandwidth-bound)
  cvt_pad_kernel<<<2048, WG, 0, stream>>>(E16, embed, 100001L, 200, 224, 100001L * 224);
  cvt_pad_kernel<<<64, WG, 0, stream>>>(W16w, wihw, 300L, 200, 224, 320L * 224);
  cvt_pad_kernel<<<32, WG, 0, stream>>>(W16s, wihs, 300L, 100, 128, 320L * 128);

  // Word level
  proj_kernel<200, 224><<<1500, WG, 0, stream>>>(GIw, E16, ids, 1500, W16w, bihw);
  bigru_kernel<<<2, WG, 0, stream>>>(foutw, GIw, whhw, bhhw, 30, 50);
  scores_kernel<<<(96000 + WG - 1) / WG, WG, 0, stream>>>(scw, foutw, wWT, wB, wC, 96000);
  attnsum_kernel<<<30, WG, 0, stream>>>(sseq, foutw, scw, 50);
  // Sentence level
  cvt_pad_kernel<<<32, WG, 0, stream>>>(S16, sseq, 1920L, 100, 128, 1920L * 128);
  proj_kernel<100, 128><<<30, WG, 0, stream>>>(GIs, S16, nullptr, 0, W16s, bihs);
  bigru_kernel<<<2, WG, 0, stream>>>(fouts, GIs, whhs, bhhs, 1, 30);
  scores_kernel<<<(1920 + WG - 1) / WG, WG, 0, stream>>>(scs, fouts, sWT, sBi, sC, 1920);
  attnsum_kernel<<<1, WG, 0, stream>>>(doc, fouts, scs, 30);
  fc_kernel<<<1, WG, 0, stream>>>(out, doc, fcw, fcb);
}
